// HarmonicSynthesizer_19138374271713
// MI455X (gfx1250) — matrix operand order repaired and checked by reading, not compiled, not run on hardware
//
#include <hip/hip_runtime.h>
#include <hip/hip_bf16.h>

typedef __attribute__((ext_vector_type(16))) _Float16 v16h;
typedef __attribute__((ext_vector_type(8)))  _Float16 v8h;
typedef __attribute__((ext_vector_type(8)))  float    v8f;

#define SRATE    48000.0f
#define TLEN     62400
#define NFRAMES  250
#define NHARM    64
#define NTILES   (TLEN / 16)          // 3900
#define WAVES_PB 8                    // 256 threads = 8 wave32
#define NCHUNK   32                   // time-chunk blocks per batch

// audio[b,t] = clip( sum_h clip(interp(X)[b,t,h],0,2) * sin((t+1)*dphi[b,h]), +-10 )
// dphi[b,h] = 2*pi*min(f0[b]*(h+1), SR/2)/SR   (freqs are frame-constant -> cumsum is analytic)
//
// Per 16-t tile: D[t,n] = sum_h sin((t+1)dphi[h]) * X[f0w+n,h]  == 16x16xK=64 GEMM
// done as two chained v_wmma_f32_16x16x32_f16, then 2-tap interp of D columns.
__global__ __launch_bounds__(256) void harmonic_synth_wmma(
    const float* __restrict__ f0_hz,
    const float* __restrict__ amplitudes,   // [B, 250, 1]
    const float* __restrict__ harm_dist,    // [B, 250, 64]
    float* __restrict__ out)                // [B, 62400]
{
    __shared__ __align__(16) _Float16 sX[NFRAMES * NHARM];   // 32 KB, f16 amp*dist
    __shared__ double sDrev[NHARM];                          // dphi in revolutions (double)
    __shared__ float  sD[WAVES_PB][256];                     // per-wave 16x16 D spill

    const int b   = blockIdx.x;
    const int tid = threadIdx.x;
    const float f0 = f0_hz[b];

    if (tid < NHARM) {
        float fr = fminf(f0 * (float)(tid + 1), SRATE * 0.5f);   // Nyquist clip
        sDrev[tid] = (double)fr / (double)SRATE;                 // phase per sample, in revs
    }
    const float* ampb = amplitudes + b * NFRAMES;
    const float* hdb  = harm_dist  + (size_t)b * NFRAMES * NHARM;
    for (int idx = tid; idx < NFRAMES * NHARM; idx += 256) {
        int f = idx >> 6;
        float x = ampb[f] * hdb[idx];
        x = fminf(fmaxf(x, 0.0f), 2.0f);                         // clip(.,0,2) (inert but kept)
        sX[idx] = (_Float16)x;
    }
    __syncthreads();

    const int wave   = tid >> 5;
    const int lane   = tid & 31;
    const int gwave  = blockIdx.y * WAVES_PB + wave;
    const int nwaves = gridDim.y * WAVES_PB;
    const float scale = (float)NFRAMES / (float)TLEN;

    const int mrow  = lane & 15;                 // A row (t in tile) / B col (frame in window)
    const int abase = (lane < 16) ? 0 : 8;       // A-matrix 16x32 f16 K-base per ISA table
    const int bkoff = (lane < 16) ? 0 : 8;       // B-matrix 32x16 f16 K-base per ISA table

    for (int tile = gwave; tile < NTILES; tile += nwaves) {
        const int tbase = tile * 16;

        // 16-frame window: covers {lo, lo+1} for all 16 samples of the tile
        float pos0 = ((float)tbase + 0.5f) * scale - 0.5f;
        int f0w = (int)floorf(pos0);
        f0w = min(max(f0w, 0), NFRAMES - 16);

        // ---- A: sin((t+1)*dphi), f16. Per lane t is fixed (row), K varies. ----
        const int n1 = tbase + mrow + 1;
        v16h a0, a1;
        #pragma unroll
        for (int i = 0; i < 16; i++) {
            int k = abase + ((i < 8) ? i : (i + 8));   // K-slot mapping (ISA 16-bit A 16x32)
            double p0 = (double)n1 * sDrev[k];       p0 -= floor(p0);   // exact phase frac
            double p1 = (double)n1 * sDrev[k + 32];  p1 -= floor(p1);
            a0[i] = (_Float16)__sinf((float)p0 * 6.2831853071795864f);
            a1[i] = (_Float16)__sinf((float)p1 * 6.2831853071795864f);
        }

        // ---- B: X^T window from LDS. Lane column = frame f0w+mrow, K contiguous. ----
        const _Float16* row = sX + (f0w + mrow) * NHARM;
        v8h b0a = *(const v8h*)(row + bkoff);
        v8h b0b = *(const v8h*)(row + bkoff + 16);
        v8h b1a = *(const v8h*)(row + bkoff + 32);
        v8h b1b = *(const v8h*)(row + bkoff + 48);
        v16h b0, b1;
        #pragma unroll
        for (int i = 0; i < 8; i++) {
            b0[i] = b0a[i]; b0[i + 8] = b0b[i];
            b1[i] = b1a[i]; b1[i + 8] = b1b[i];
        }

        // ---- D = A x B, K=64 via two chained WMMAs (f32 accumulate) ----
        v8f c = {};
        c = __builtin_amdgcn_wmma_f32_16x16x32_f16(false, a0, false, b0, (short)0, c, false, false);
        c = __builtin_amdgcn_wmma_f32_16x16x32_f16(false, a1, false, b1, (short)0, c, false, false);

        // ---- spill D[m][n] to this wave's LDS slab (n = lane%16, m = r + 8*(lane/16)) ----
        const int mhalf = (lane >> 4) << 3;
        #pragma unroll
        for (int r = 0; r < 8; r++)
            sD[wave][(r + mhalf) * 16 + mrow] = c[r];
        __asm__ volatile("s_wait_dscnt 0" ::: "memory");   // wave-local LDS RAW fence

        // ---- 2-tap interp across frames + final clip, 16 outputs per wave ----
        if (lane < 16) {
            int t = tbase + lane;
            float pos = ((float)t + 0.5f) * scale - 0.5f;
            pos = fminf(fmaxf(pos, 0.0f), (float)(NFRAMES - 1));
            int lo = (int)floorf(pos);
            float w = pos - (float)lo;
            int hi = min(lo + 1, NFRAMES - 1);
            float d0 = sD[wave][lane * 16 + (lo - f0w)];
            float d1 = sD[wave][lane * 16 + (hi - f0w)];
            float audio = (1.0f - w) * d0 + w * d1;
            audio = fminf(fmaxf(audio, -10.0f), 10.0f);
            out[(size_t)b * TLEN + t] = audio;
        }
    }
}

extern "C" void kernel_launch(void* const* d_in, const int* in_sizes, int n_in,
                              void* d_out, int out_size, void* d_ws, size_t ws_size,
                              hipStream_t stream) {
    const float* f0  = (const float*)d_in[0];   // [16]
    const float* amp = (const float*)d_in[1];   // [16,250,1]
    const float* hd  = (const float*)d_in[2];   // [16,250,64]
    float* out = (float*)d_out;                 // [16,62400]

    const int batch = in_sizes[0];              // 16
    dim3 grid(batch, NCHUNK);
    harmonic_synth_wmma<<<grid, 256, 0, stream>>>(f0, amp, hd, out);
}
